// FlowNet_68152541053332
// MI455X (gfx1250) — matrix operand order repaired and checked by reading, not compiled, not run on hardware
//
#include <hip/hip_runtime.h>
#include <hip/hip_bf16.h>

typedef _Float16 v16h __attribute__((ext_vector_type(16)));
typedef _Float16 v8h  __attribute__((ext_vector_type(8)));
typedef float    v8f  __attribute__((ext_vector_type(8)));

#define EPSF 1e-6f

static constexpr int BB = 2, CF = 32, DD = 48, HH = 48, WW = 48;
static constexpr int DHW  = DD * HH * WW;   // 110592
static constexpr int NVOX = BB * DHW;       // 221184

// ---------------------------------------------------------------- utilities
__global__ void k_zero(float* p, int n) {
  int i = blockIdx.x * blockDim.x + threadIdx.x;
  if (i < n) p[i] = 0.f;
}

// NCDHW f32 -> channels-last f32 (C = 32)
__global__ void k_nchw_to_cl(const float* __restrict__ in, float* __restrict__ out) {
  int i = blockIdx.x * blockDim.x + threadIdx.x;
  if (i >= NVOX * CF) return;
  int v = i >> 5;            // global voxel (includes batch)
  int c = i & 31;
  int b = v / DHW;
  int vl = v - b * DHW;
  out[(size_t)v * CF + c] = in[(size_t)(b * CF + c) * DHW + vl];
}

// weights (Cout=64, Cin, 3,3,3) f32 -> per-tap packed f16 [tap][o][ci_pad]
__global__ void k_pack_w(const float* __restrict__ w, _Float16* __restrict__ wp,
                         int cin, int cpad) {
  int i = blockIdx.x * blockDim.x + threadIdx.x;
  int total = 27 * 64 * cpad;
  if (i >= total) return;
  int t = i / (64 * cpad);
  int r = i - t * 64 * cpad;
  int o = r / cpad;
  int ci = r - o * cpad;
  float v = (ci < cin) ? w[((size_t)(o * cin + ci)) * 27 + t] : 0.f;
  wp[((size_t)t * 64 + o) * cpad + ci] = (_Float16)v;
}

// trilinear resize of channels-last feats (48^3 -> d^3), half-pixel centers
__global__ void k_resize_cl(const float* __restrict__ in, float* __restrict__ out,
                            int d, int hgt, int wid) {
  int i = blockIdx.x * blockDim.x + threadIdx.x;
  int dhw_ = d * hgt * wid;
  int nv = BB * dhw_;
  if (i >= nv) return;
  int b = i / dhw_;
  int r = i - b * dhw_;
  int z = r / (hgt * wid); r -= z * hgt * wid;
  int y = r / wid;
  int x = r - y * wid;
  float fz = (z + 0.5f) * ((float)DD / d) - 0.5f;
  float fy = (y + 0.5f) * ((float)HH / hgt) - 0.5f;
  float fx = (x + 0.5f) * ((float)WW / wid) - 0.5f;
  int z0 = (int)floorf(fz), y0 = (int)floorf(fy), x0 = (int)floorf(fx);
  float wz = fz - z0, wy = fy - y0, wx = fx - x0;
  float o0[CF];
  for (int c = 0; c < CF; ++c) o0[c] = 0.f;
  for (int iz = 0; iz < 2; ++iz)
    for (int iy = 0; iy < 2; ++iy)
      for (int ix = 0; ix < 2; ++ix) {
        int zc = min(max(z0 + iz, 0), DD - 1);
        int yc = min(max(y0 + iy, 0), HH - 1);
        int xc = min(max(x0 + ix, 0), WW - 1);
        float wgt = (iz ? wz : 1.f - wz) * (iy ? wy : 1.f - wy) * (ix ? wx : 1.f - wx);
        const float* src = in + ((size_t)((b * DD + zc) * HH + yc) * WW + xc) * CF;
        for (int c = 0; c < CF; c += 4) {
          float4 f = *(const float4*)(src + c);
          o0[c] += wgt * f.x; o0[c + 1] += wgt * f.y;
          o0[c + 2] += wgt * f.z; o0[c + 3] += wgt * f.w;
        }
      }
  float* dst = out + (size_t)i * CF;
  for (int c = 0; c < CF; ++c) dst[c] = o0[c];
}

// 27-tap correlation + relu + L2 normalize over taps; channels-last f16 (pad 27->32)
__global__ void k_correlate(const float* __restrict__ f0, const float* __restrict__ f1,
                            _Float16* __restrict__ cc, int d, int hgt, int wid) {
  int i = blockIdx.x * blockDim.x + threadIdx.x;
  int dhw_ = d * hgt * wid;
  int nv = BB * dhw_;
  if (i >= nv) return;
  int b = i / dhw_;
  int r = i - b * dhw_;
  int z = r / (hgt * wid); r -= z * hgt * wid;
  int y = r / wid;
  int x = r - y * wid;
  float a[CF];
  const float* p0 = f0 + (size_t)i * CF;
  for (int c = 0; c < CF; ++c) a[c] = p0[c];
  float vals[27];
  float ssq = 0.f;
  for (int t = 0; t < 27; ++t) {
    int dz = t / 9 - 1, dy = (t / 3) % 3 - 1, dx = t % 3 - 1;
    int zz = z + dz, yy = y + dy, xx = x + dx;
    float vdot = 0.f;
    if (zz >= 0 && zz < d && yy >= 0 && yy < hgt && xx >= 0 && xx < wid) {
      const float* p1 = f1 + ((size_t)((b * d + zz) * hgt + yy) * wid + xx) * CF;
      for (int c = 0; c < CF; ++c) vdot += a[c] * p1[c];
    }
    vdot = fmaxf(vdot, 0.f);      // relu
    vals[t] = vdot;
    ssq += vdot * vdot;
  }
  float inv = 1.f / fmaxf(sqrtf(ssq), 1e-12f);
  _Float16* dst = cc + (size_t)i * 32;
  for (int t = 0; t < 27; ++t) dst[t] = (_Float16)(vals[t] * inv);
  for (int t = 27; t < 32; ++t) dst[t] = (_Float16)0.f;
}

// ------------------------------------------------- implicit-GEMM conv via WMMA
// Block = 256 threads (8 wave32) -> 64 outch x 32 voxel tile; no LDS, no barriers.
// DV (spatial dim) is a template parameter so the voxel decompose is mul-shift,
// not runtime integer division. Both WMMA fragments load straight from
// (L2/WGP$-resident) global memory:
//   A (weights): per-lane two 16B spans of per-tap packed wpack[t][m][k],
//     matching the 16-bit A VGPR layout (K pairs 0-7/16-23 vs 8-15/24-31).
//   B (im2col): lane n's own voxel's channels-last span, same K pairs as A per
//     lane half (koffB); halo voxels exec-masked to zero (WMMA runs EXEC=all-1).
// #pragma unroll 9 keeps ~9 taps of global_load_b128 in flight per wave.
template <int CIN_PAD, bool LEAKY, int DV>
__global__ void __launch_bounds__(256)
k_conv_wmma(const _Float16* __restrict__ in_cl, const _Float16* __restrict__ wpack,
            const float* __restrict__ bias, _Float16* __restrict__ out_cl) {
  constexpr int dhw_ = DV * DV * DV;
  constexpr int nvox = BB * dhw_;
  const int tid   = threadIdx.x;
  const int lane  = tid & 31;
  const int wave  = tid >> 5;
  const int mtile = wave & 3;
  const int ntile = wave >> 2;
  const int vb    = blockIdx.x * 32;
  const int m16   = lane & 15;
  const int hoffA = (lane >> 4) << 3;   // A: K base 0 or 8
  const int koffB = (lane >> 4) << 3;   // B: K base 0 or 8

  // this lane's B-matrix row == its output voxel (constant-divisor decompose)
  const int nvx = vb + ntile * 16 + m16;
  const int b = nvx / dhw_;
  int r = nvx - b * dhw_;
  const int z = r / (DV * DV); r -= z * DV * DV;
  const int y = r / DV;
  const int x = r - y * DV;

  const _Float16* arow = wpack + (size_t)(mtile * 16 + m16) * CIN_PAD + hoffA;

  v8f acc = {};
#pragma unroll 9
  for (int t = 0; t < 27; ++t) {
    const int dz = t / 9 - 1, dy = (t / 3) % 3 - 1, dx = t % 3 - 1;
    const int zz = z + dz, yy = y + dy, xx = x + dx;
    const bool ok = zz >= 0 && zz < DV && yy >= 0 && yy < DV && xx >= 0 && xx < DV;
    const _Float16* ap0 = arow + (size_t)t * 64 * CIN_PAD;
    const _Float16* brow =
        in_cl + ((size_t)((b * DV + zz) * DV + yy) * DV + xx) * CIN_PAD + koffB;
    if (t < 26)
      __builtin_prefetch(arow + (size_t)(t + 1) * 64 * CIN_PAD, 0, 1);
#pragma unroll
    for (int ks = 0; ks < CIN_PAD / 32; ++ks) {
      const _Float16* ap = ap0 + ks * 32;
      v16h av, bv;
      ((v8h*)&av)[0] = *(const v8h*)ap;          // K base .. base+7
      ((v8h*)&av)[1] = *(const v8h*)(ap + 16);   // K base+16 .. base+23
      v8h b0 = {}, b1 = {};
      if (ok) {
        b0 = *(const v8h*)(brow + ks * 32);
        b1 = *(const v8h*)(brow + ks * 32 + 16);
      }
      ((v8h*)&bv)[0] = b0;
      ((v8h*)&bv)[1] = b1;
      acc = __builtin_amdgcn_wmma_f32_16x16x32_f16(false, av, false, bv, (short)0, acc,
                                                   false, false);
    }
  }
  // epilogue: bias + leaky relu, write channels-last f16
  const int mbase = mtile * 16 + hoffA;
  if (nvx < nvox) {
#pragma unroll
    for (int r2 = 0; r2 < 8; ++r2) {
      int m = mbase + r2;
      float v = acc[r2] + bias[m];
      if (LEAKY) v = v > 0.f ? v : 0.1f * v;
      out_cl[(size_t)nvx * 64 + m] = (_Float16)v;
    }
  }
}

// 1x1x1 conv 64->3, writes flow in NCDHW f32
__global__ void k_conv3(const _Float16* __restrict__ hid, const float* __restrict__ w3,
                        const float* __restrict__ b3, float* __restrict__ flow,
                        int d, int hgt, int wid) {
  int i = blockIdx.x * blockDim.x + threadIdx.x;
  int dhw_ = d * hgt * wid, nv = BB * dhw_;
  if (i >= nv) return;
  int b = i / dhw_;
  int vl = i - b * dhw_;
  const _Float16* hp = hid + (size_t)i * 64;
  float h0[64];
  for (int c = 0; c < 64; ++c) h0[c] = (float)hp[c];
  for (int o = 0; o < 3; ++o) {
    float s = b3[o];
    const float* wr = w3 + o * 64;
    for (int c = 0; c < 64; ++c) s += wr[c] * h0[c];
    flow[(size_t)(b * 3 + o) * dhw_ + vl] = s;
  }
}

// flow_total += resize(flow_sc, 48^3) / sc   (identity when d == 48)
__global__ void k_flow_upadd(const float* __restrict__ fs, float* __restrict__ ft,
                             int d, int hgt, int wid, float inv_sc) {
  int i = blockIdx.x * blockDim.x + threadIdx.x;
  if (i >= BB * 3 * DHW) return;
  int bc = i / DHW;
  int r = i - bc * DHW;
  int z = r / (HH * WW); r -= z * HH * WW;
  int y = r / WW;
  int x = r - y * WW;
  float fz = (z + 0.5f) * ((float)d / DD) - 0.5f;
  float fy = (y + 0.5f) * ((float)hgt / HH) - 0.5f;
  float fx = (x + 0.5f) * ((float)wid / WW) - 0.5f;
  int z0 = (int)floorf(fz), y0 = (int)floorf(fy), x0 = (int)floorf(fx);
  float wz = fz - z0, wy = fy - y0, wx = fx - x0;
  const float* src = fs + (size_t)bc * d * hgt * wid;
  float s = 0.f;
  for (int iz = 0; iz < 2; ++iz)
    for (int iy = 0; iy < 2; ++iy)
      for (int ix = 0; ix < 2; ++ix) {
        int zc = min(max(z0 + iz, 0), d - 1);
        int yc = min(max(y0 + iy, 0), hgt - 1);
        int xc = min(max(x0 + ix, 0), wid - 1);
        float wgt = (iz ? wz : 1.f - wz) * (iy ? wy : 1.f - wy) * (ix ? wx : 1.f - wx);
        s += wgt * src[((size_t)zc * hgt + yc) * wid + xc];
      }
  ft[i] += s * inv_sc;
}

// backwarp feat1 with flow_total (zeros outside), plus warp-loss reduction on last scale
__global__ void __launch_bounds__(256)
k_backwarp(const float* __restrict__ f1cl, const float* __restrict__ ftot,
           const float* __restrict__ f0cl, float* __restrict__ out_cl,
           float* __restrict__ accum, int do_loss) {
  __shared__ float red0[256], red1[256];
  int i = blockIdx.x * blockDim.x + threadIdx.x;
  float lnum = 0.f, lden = 0.f;
  if (i < NVOX) {
    int b = i / DHW;
    int vl = i - b * DHW;
    int rr = vl;
    int z = rr / (HH * WW); rr -= z * HH * WW;
    int y = rr / WW;
    int x = rr - y * WW;
    float fx = x + ftot[(size_t)(b * 3 + 0) * DHW + vl];
    float fy = y + ftot[(size_t)(b * 3 + 1) * DHW + vl];
    float fz = z + ftot[(size_t)(b * 3 + 2) * DHW + vl];
    int x0 = (int)floorf(fx), y0 = (int)floorf(fy), z0 = (int)floorf(fz);
    float o0[CF];
    for (int c = 0; c < CF; ++c) o0[c] = 0.f;
    float vs = 0.f;
    for (int iz = 0; iz < 2; ++iz)
      for (int iy = 0; iy < 2; ++iy)
        for (int ix = 0; ix < 2; ++ix) {
          int xc = x0 + ix, yc = y0 + iy, zc = z0 + iz;
          float wgt = (1.f - fabsf(fx - xc)) * (1.f - fabsf(fy - yc)) *
                      (1.f - fabsf(fz - zc));
          bool valid = xc >= 0 && xc < WW && yc >= 0 && yc < HH && zc >= 0 && zc < DD;
          if (valid) {
            vs += wgt;
            const float* s = f1cl + ((size_t)((b * DD + zc) * HH + yc) * WW + xc) * CF;
            for (int c = 0; c < CF; ++c) o0[c] += wgt * s[c];
          }
        }
    float* dst = out_cl + (size_t)i * CF;
    for (int c = 0; c < CF; ++c) dst[c] = o0[c];
    if (do_loss) {
      const float* p0 = f0cl + (size_t)i * CF;
      float ss = 0.f;
      for (int c = 0; c < CF; ++c) {
        float dd = o0[c] - p0[c];
        ss += dd * dd;
      }
      lnum = sqrtf(EPSF + ss) * vs;
      lden = vs;
    }
  }
  if (do_loss) {
    red0[threadIdx.x] = lnum;
    red1[threadIdx.x] = lden;
    __syncthreads();
    for (int s = 128; s > 0; s >>= 1) {
      if (threadIdx.x < (unsigned)s) {
        red0[threadIdx.x] += red0[threadIdx.x + s];
        red1[threadIdx.x] += red1[threadIdx.x + s];
      }
      __syncthreads();
    }
    if (threadIdx.x == 0) {
      atomicAdd(&accum[0], red0[0]);
      atomicAdd(&accum[1], red1[0]);
    }
  }
}

// final losses over flow_total + copy flow to output
__global__ void __launch_bounds__(256)
k_loss(const float* __restrict__ ft, const float* __restrict__ fg,
       const float* __restrict__ mg, float* __restrict__ accum,
       float* __restrict__ out_flow) {
  __shared__ float r0[256], r1[256], r2[256], r3[256];
  int i = blockIdx.x * blockDim.x + threadIdx.x;
  float a_l1 = 0.f, a_l2 = 0.f, a_m = 0.f, a_sm = 0.f;
  if (i < NVOX) {
    int b = i / DHW;
    int vl = i - b * DHW;
    int rr = vl;
    int z = rr / (HH * WW); rr -= z * HH * WW;
    int y = rr / WW;
    int x = rr - y * WW;
    (void)z;
    float m = mg[i];
    float l1 = 0.f, l2 = 0.f, sm = 0.f;
    for (int c = 0; c < 3; ++c) {
      size_t idx = (size_t)(b * 3 + c) * DHW + vl;
      float f = ft[idx];
      out_flow[idx] = f;
      float df = f - fg[idx];
      float ad = fabsf(df);
      l1 += (ad < 1.f) ? 0.5f * df * df : ad - 0.5f;   // SmoothL1, beta=1
      l2 += df * df;
      float dxv = (x < WW - 1) ? fabsf(ft[idx + 1] - f) : 0.f;
      float dyv = (y < HH - 1) ? fabsf(ft[idx + WW] - f) : 0.f;
      sm += 2.f * dxv + dyv;   // reference sums dx + dy + dx (dz unused)
    }
    a_l1 = (l1 / 3.f) * m;
    a_l2 = (l2 / 3.f) * m;
    a_m = m;
    a_sm = sm;
  }
  r0[threadIdx.x] = a_l1; r1[threadIdx.x] = a_l2;
  r2[threadIdx.x] = a_m;  r3[threadIdx.x] = a_sm;
  __syncthreads();
  for (int s = 128; s > 0; s >>= 1) {
    if (threadIdx.x < (unsigned)s) {
      r0[threadIdx.x] += r0[threadIdx.x + s];
      r1[threadIdx.x] += r1[threadIdx.x + s];
      r2[threadIdx.x] += r2[threadIdx.x + s];
      r3[threadIdx.x] += r3[threadIdx.x + s];
    }
    __syncthreads();
  }
  if (threadIdx.x == 0) {
    atomicAdd(&accum[2], r0[0]);
    atomicAdd(&accum[3], r1[0]);
    atomicAdd(&accum[4], r2[0]);
    atomicAdd(&accum[5], r3[0]);
  }
}

__global__ void k_final(const float* __restrict__ accum, float* __restrict__ out) {
  if (threadIdx.x == 0 && blockIdx.x == 0) {
    float warp = accum[0] / (accum[1] + EPSF);
    float msum = accum[4];
    float l1 = accum[2] / (msum + EPSF);
    float l2 = accum[3] / (msum + EPSF);
    float sm = accum[5] / (float)(BB * 3 * DHW);
    out[0] = l1 + l2 + warp + 0.1f * sm;
  }
}

// -------------------------------------------------------------------- driver
template <int DV>
static void launch_convs(const _Float16* cc16, const _Float16* wp1, const float* b1,
                         _Float16* hid1, const _Float16* wp2, const float* b2,
                         _Float16* hid2, hipStream_t stream) {
  const int nv = BB * DV * DV * DV;
  k_conv_wmma<32, true, DV><<<nv / 32, 256, 0, stream>>>(cc16, wp1, b1, hid1);
  k_conv_wmma<64, true, DV><<<nv / 32, 256, 0, stream>>>(hid1, wp2, b2, hid2);
}

extern "C" void kernel_launch(void* const* d_in, const int* in_sizes, int n_in,
                              void* d_out, int out_size, void* d_ws, size_t ws_size,
                              hipStream_t stream) {
  const float* feat0  = (const float*)d_in[0];
  const float* feat1  = (const float*)d_in[1];
  const float* flow_g = (const float*)d_in[2];
  const float* mask_g = (const float*)d_in[3];
  const float* w1 = (const float*)d_in[4];
  const float* b1 = (const float*)d_in[5];
  const float* w2 = (const float*)d_in[6];
  const float* b2 = (const float*)d_in[7];
  const float* w3 = (const float*)d_in[8];
  const float* b3 = (const float*)d_in[9];
  float* out = (float*)d_out;

  char* ws = (char*)d_ws;
  size_t off = 0;
  auto carve = [&](size_t bytes) -> void* {
    void* p = ws + off;
    off = (off + bytes + 255) & ~(size_t)255;
    return p;
  };
  float*    feat0_cl = (float*)carve((size_t)NVOX * CF * 4);
  float*    feat1_cl = (float*)carve((size_t)NVOX * CF * 4);
  float*    feat1_al = (float*)carve((size_t)NVOX * CF * 4);
  float*    feat0_s  = (float*)carve((size_t)NVOX * CF * 4);
  float*    feat1_s  = (float*)carve((size_t)NVOX * CF * 4);
  _Float16* cc16     = (_Float16*)carve((size_t)NVOX * 32 * 2);
  _Float16* hid1     = (_Float16*)carve((size_t)NVOX * 64 * 2);
  _Float16* hid2     = (_Float16*)carve((size_t)NVOX * 64 * 2);
  float*    flow_sc  = (float*)carve((size_t)NVOX * 3 * 4);
  float*    flow_tot = (float*)carve((size_t)NVOX * 3 * 4);
  _Float16* wp1      = (_Float16*)carve((size_t)27 * 64 * 32 * 2);
  _Float16* wp2      = (_Float16*)carve((size_t)27 * 64 * 64 * 2);
  float*    accum    = (float*)carve(32 * 4);

  auto nb = [](int n) { return (n + 255) / 256; };

  k_zero<<<nb(NVOX * 3), 256, 0, stream>>>(flow_tot, NVOX * 3);
  k_zero<<<1, 32, 0, stream>>>(accum, 8);
  k_nchw_to_cl<<<nb(NVOX * CF), 256, 0, stream>>>(feat0, feat0_cl);
  k_nchw_to_cl<<<nb(NVOX * CF), 256, 0, stream>>>(feat1, feat1_cl);
  k_pack_w<<<nb(27 * 64 * 32), 256, 0, stream>>>(w1, wp1, 27, 32);
  k_pack_w<<<nb(27 * 64 * 64), 256, 0, stream>>>(w2, wp2, 64, 64);

  const int   dims[4] = {12, 24, 36, 48};
  const float scs[4]  = {0.25f, 0.5f, 0.75f, 1.0f};
  for (int si = 0; si < 4; ++si) {
    const int d = dims[si];
    const int nv = BB * d * d * d;
    const float* f1src = (si == 0) ? feat1_cl : feat1_al;
    const float* f0u;
    const float* f1u;
    if (d != DD) {
      k_resize_cl<<<nb(nv), 256, 0, stream>>>(feat0_cl, feat0_s, d, d, d);
      k_resize_cl<<<nb(nv), 256, 0, stream>>>(f1src, feat1_s, d, d, d);
      f0u = feat0_s;
      f1u = feat1_s;
    } else {
      f0u = feat0_cl;
      f1u = f1src;
    }
    k_correlate<<<nb(nv), 256, 0, stream>>>(f0u, f1u, cc16, d, d, d);
    switch (d) {
      case 12: launch_convs<12>(cc16, wp1, b1, hid1, wp2, b2, hid2, stream); break;
      case 24: launch_convs<24>(cc16, wp1, b1, hid1, wp2, b2, hid2, stream); break;
      case 36: launch_convs<36>(cc16, wp1, b1, hid1, wp2, b2, hid2, stream); break;
      default: launch_convs<48>(cc16, wp1, b1, hid1, wp2, b2, hid2, stream); break;
    }
    k_conv3<<<nb(nv), 256, 0, stream>>>(hid2, w3, b3, flow_sc, d, d, d);
    k_flow_upadd<<<nb(NVOX * 3), 256, 0, stream>>>(flow_sc, flow_tot, d, d, d,
                                                   1.f / scs[si]);
    k_backwarp<<<nb(NVOX), 256, 0, stream>>>(feat1_cl, flow_tot, feat0_cl, feat1_al,
                                             accum, (si == 3) ? 1 : 0);
  }
  k_loss<<<nb(NVOX), 256, 0, stream>>>(flow_tot, flow_g, mask_g, accum, out + 1);
  k_final<<<1, 64, 0, stream>>>(accum, out);
}
